// LipschitzNetWithBatches_15745350107400
// MI455X (gfx1250) — matrix operand order repaired and checked by reading, not compiled, not run on hardware
//
#include <hip/hip_runtime.h>
#include <math.h>

// ---------------------------------------------------------------------------
// Lipschitz RNN on MI455X (gfx1250):
//   h <- h + DT*(h@A) + DT*tanh(h@W + x_t@Wu + bu),  T=2048 steps
//   B=256, HID=512, IN=64, OUT=64.
// Weights register-resident: 8 teams x 4 workgroups; each WG owns 128 hidden
// columns (256KB bf16 weights = 128 VGPRs/lane over 16 waves). Per-step team
// exchange of h through an L2-resident buffer + agent-scope atomic barrier.
// ---------------------------------------------------------------------------

typedef __bf16 bf16;
typedef __attribute__((ext_vector_type(16))) __bf16 v16bf;
typedef __attribute__((ext_vector_type(8)))  __bf16 v8bf;
typedef __attribute__((ext_vector_type(8)))  float  v8f;

#define HID     512
#define IND     64
#define TT      2048
#define BATCH   256
#define TB      32                 // batch rows per team
#define TEAM_WG 4                  // workgroups per team (128 cols each)
#define NTEAM   (BATCH / TB)       // 8
#define NWG     (NTEAM * TEAM_WG)  // 32
#define NTHR    512                // 16 waves: 8 col-tiles x 2 K-halves
#define HB_S    520                // LDS h stride (bf16): 4m%64 banks, conflict-free
#define XB_S    72                 // LDS x stride (bf16)

#define DT_C    0.001f
#define GAMMA_C 0.001f
#define CBETA   (-0.5f)            // (1-beta)-beta, beta=0.75

// ---- WMMA fragment loaders (contiguous k = 16*(lane/16)+s, same for A and B) ----------------------

// A 16x32 bf16: lane row m=lane&15 (+16*mt); elems[0..15]=K k0+(lane>=16?16:0)..+15.
__device__ __forceinline__ v16bf load_a_frag(const bf16* base, int mt, int k0,
                                             int stride, int lane) {
    int m  = (lane & 15) + (mt << 4);
    int kb = k0 + ((lane >> 4) << 4);
    const bf16* p = base + m * stride + kb;
    v8bf lo = *(const v8bf*)(p);
    v8bf hi = *(const v8bf*)(p + 8);
    return __builtin_shufflevector(lo, hi, 0,1,2,3,4,5,6,7,8,9,10,11,12,13,14,15);
}

// B 32x16 bf16, weights stored column-major (mat[n*rowlen+k]): lane col n=lane&15,
// elems[0..15] = K = k0+(lane>=16?16:0)..+15 (one 32B contiguous load).
__device__ __forceinline__ v16bf load_b_frag(const bf16* __restrict__ mat,
                                             int col, int k0, int rowlen, int lane) {
    int kb = k0 + ((lane >> 4) << 4);
    return *(const v16bf*)(mat + col * rowlen + kb);
}

#define WMMA_BF16(a, b, c) \
    __builtin_amdgcn_wmma_f32_16x16x32_bf16(false, (a), false, (b), (short)0, (c), false, false)

// tanh without a libm call (avoid spilling 128 weight VGPRs around a call):
// tanh(|x|) = 1 - 2/(exp2(2*log2e*|x|)+1), sign restored with copysign.
__device__ __forceinline__ float fast_tanh(float x) {
    float ax = fabsf(x);
    float e  = __builtin_amdgcn_exp2f(ax * 2.8853900817779268f); // 2*log2(e)
    float r  = 1.0f - 2.0f * __builtin_amdgcn_rcpf(e + 1.0f);
    return copysignf(r, x);
}

// ---------------------------------------------------------------------------
// Prep: A = M_A + CBETA*M_A^T - gamma*I (W likewise), stored transposed bf16;
// Wu transposed bf16; h buffers = 1.0; team barrier counters = 0 (every call,
// keeping graph replays deterministic).
// ---------------------------------------------------------------------------
__global__ void prep_mats(const float* __restrict__ MW, const float* __restrict__ MA,
                          const float* __restrict__ Wu,
                          bf16* __restrict__ At, bf16* __restrict__ Wt,
                          bf16* __restrict__ Wut, bf16* __restrict__ hbuf,
                          unsigned* __restrict__ barc) {
    int idx = blockIdx.x * blockDim.x + threadIdx.x;
    if (idx < HID * HID) {
        int n = idx >> 9, k = idx & (HID - 1);
        float diag = (n == k) ? GAMMA_C : 0.0f;
        At[n * HID + k] = (bf16)(MA[k * HID + n] + CBETA * MA[n * HID + k] - diag);
        Wt[n * HID + k] = (bf16)(MW[k * HID + n] + CBETA * MW[n * HID + k] - diag);
    }
    if (idx < HID * IND) {
        int h = idx >> 6, i = idx & (IND - 1);
        Wut[h * IND + i] = (bf16)(Wu[i * HID + h]);
    }
    if (idx < NTEAM * TB * HID) hbuf[idx] = (bf16)1.0f;
    if (idx < NTEAM * 16)       barc[idx] = 0u;
}

// ---------------------------------------------------------------------------
__global__ __launch_bounds__(NTHR, 1) void lipschitz_rnn(
        const float* __restrict__ x,    // [B, T, IND]
        const float* __restrict__ bu,   // [HID]
        const float* __restrict__ Wd,   // [HID, 64] row-major
        const bf16*  __restrict__ At,   // [HID, HID] col-major bf16
        const bf16*  __restrict__ Wt,   // [HID, HID] col-major bf16
        const bf16*  __restrict__ Wut,  // [HID, IND] col-major bf16
        bf16*  __restrict__ hbuf,       // [NTEAM][TB][HID] bf16 exchange buffer
        unsigned* __restrict__ barc,    // [NTEAM*16] barrier counters
        float* __restrict__ out)        // [B, 64]
{
    __shared__ __align__(16) bf16  h_bf[TB * HB_S];            // 33,280 B
    __shared__ __align__(16) bf16  x_bf[TB * XB_S];            //  4,608 B
    __shared__ __align__(32) float scr[8 * 2 * 32 * 8];        // 16,384 B

    const int tid   = threadIdx.x;
    const int lane  = tid & 31;
    const int wave  = tid >> 5;
    const int team  = blockIdx.x / TEAM_WG;
    const int rank  = blockIdx.x % TEAM_WG;
    const int cg    = wave & 7;        // column tile within this WG
    const int khalf = wave >> 3;       // K-half: 0 -> K[0,256), 1 -> K[256,512)
    const int col   = rank * 128 + cg * 16 + (lane & 15);
    const int hi8   = (lane >> 4) << 3;

    bf16* hteam = hbuf + (size_t)team * (TB * HID);
    unsigned* cnt = barc + team * 16;
    const float bias = bu[col];

    // ---- register-resident weight fragments (128 VGPRs/lane) ----
    v16bf wA[8], wW[8];
#pragma unroll
    for (int kc = 0; kc < 8; ++kc) {
        int k0 = khalf * 256 + kc * 32;
        wA[kc] = load_b_frag(At, col, k0, HID, lane);
        wW[kc] = load_b_frag(Wt, col, k0, HID, lane);
    }
    v16bf wU = load_b_frag(Wut, col, khalf * 32, IND, lane);

    // f32 master copy of h (D-fragment layout), owned by low (khalf==0) waves
    float hreg[2][8];
#pragma unroll
    for (int mt = 0; mt < 2; ++mt)
#pragma unroll
        for (int r = 0; r < 8; ++r) hreg[mt][r] = 1.0f;

    for (int t = 0; t < TT; ++t) {
        // ---- stage x_t tile [TB,IND] -> LDS bf16 ----
#pragma unroll
        for (int j = 0; j < 4; ++j) {
            int e = tid + j * NTHR;
            int m = e >> 6, i = e & (IND - 1);
            x_bf[m * XB_S + i] = (bf16)x[((size_t)(team * TB + m) * TT + t) * IND + i];
        }
        // ---- stage team h [TB,HID] bf16 -> LDS (uint4 bulk copy) ----
        {
            const uint4* hsrc = (const uint4*)hteam;
#pragma unroll
            for (int j = 0; j < 4; ++j) {
                int g = tid + j * NTHR;             // uint4 index; 64 per row
                int m = g >> 6, c8 = g & 63;
                *(uint4*)&h_bf[m * HB_S + c8 * 8] = hsrc[g];
            }
        }
        __syncthreads();

        // ---- GEMMs: this wave's K-half of h@A, h@W (+ x@Wu share) ----
        v8f accA[2], accW[2];
#pragma unroll
        for (int mt = 0; mt < 2; ++mt) {
            v8f aA = {}, aW = {};
            v16bf ax = load_a_frag(x_bf, mt, khalf * 32, XB_S, lane);
            aW = WMMA_BF16(ax, wU, aW);
#pragma unroll
            for (int kc = 0; kc < 8; ++kc) {
                v16bf a = load_a_frag(h_bf, mt, khalf * 256 + kc * 32, HB_S, lane);
                aA = WMMA_BF16(a, wA[kc], aA);
                aW = WMMA_BF16(a, wW[kc], aW);
            }
            accA[mt] = aA; accW[mt] = aW;
        }

        // ---- K-half combine (two phases, 16KB LDS) + h update ----
#pragma unroll
        for (int mt = 0; mt < 2; ++mt) {
            if (khalf) {
                *(v8f*)&scr[((cg * 2 + 0) * 32 + lane) * 8] = accA[mt];
                *(v8f*)&scr[((cg * 2 + 1) * 32 + lane) * 8] = accW[mt];
            }
            __syncthreads();
            if (!khalf) {
                v8f pA = *(const v8f*)&scr[((cg * 2 + 0) * 32 + lane) * 8];
                v8f pW = *(const v8f*)&scr[((cg * 2 + 1) * 32 + lane) * 8];
#pragma unroll
                for (int r = 0; r < 8; ++r) {
                    float sA = accA[mt][r] + pA[r];
                    float sW = accW[mt][r] + pW[r] + bias;
                    float hn = hreg[mt][r] + DT_C * sA + DT_C * fast_tanh(sW);
                    hreg[mt][r] = hn;
                    int m = (mt << 4) + r + hi8;
                    hteam[m * HID + col] = (bf16)hn;   // publish bf16 slice
                }
            }
            __syncthreads();   // scr reuse / last one doubles as pre-signal sync
        }

        // ---- team barrier: release h slices, arrive, spin, acquire ----
        __threadfence();                                  // release our stores
        if (tid == 0) {
            __hip_atomic_fetch_add(cnt, 1u, __ATOMIC_RELEASE, __HIP_MEMORY_SCOPE_AGENT);
            unsigned target = (unsigned)(TEAM_WG * (t + 1));
            while (__hip_atomic_load(cnt, __ATOMIC_ACQUIRE, __HIP_MEMORY_SCOPE_AGENT) < target)
                __builtin_amdgcn_s_sleep(1);
        }
        __syncthreads();
        __threadfence();                                  // acquire: invalidate stale h
    }

    // ---- final: restage h, project out[b][o] = sum_k h[b][k]*Wd[k][o] ----
    {
        const uint4* hsrc = (const uint4*)hteam;
#pragma unroll
        for (int j = 0; j < 4; ++j) {
            int g = tid + j * NTHR;
            int m = g >> 6, c8 = g & 63;
            *(uint4*)&h_bf[m * HB_S + c8 * 8] = hsrc[g];
        }
    }
    __syncthreads();
    // each rank writes its own 16-wide slice of the 64 output columns
    for (int e = tid; e < TB * 16; e += NTHR) {
        int m = e >> 4, o = rank * 16 + (e & 15);
        float s = 0.0f;
        const bf16* hrow = h_bf + m * HB_S;
#pragma unroll 8
        for (int k = 0; k < HID; ++k)
            s += (float)hrow[k] * Wd[k * 64 + o];
        out[(size_t)(team * TB + m) * 64 + o] = s;
    }
}

// ---------------------------------------------------------------------------
extern "C" void kernel_launch(void* const* d_in, const int* in_sizes, int n_in,
                              void* d_out, int out_size, void* d_ws, size_t ws_size,
                              hipStream_t stream) {
    const float* x  = (const float*)d_in[0];
    const float* MW = (const float*)d_in[1];
    const float* MA = (const float*)d_in[2];
    const float* Wu = (const float*)d_in[3];
    const float* bu = (const float*)d_in[4];
    const float* Wd = (const float*)d_in[5];

    bf16* At   = (bf16*)d_ws;                         // 512 KB
    bf16* Wt   = At + HID * HID;                      // 512 KB
    bf16* Wut  = Wt + HID * HID;                      //  64 KB
    bf16* hbuf = Wut + HID * IND;                     // 256 KB
    unsigned* barc = (unsigned*)(hbuf + NTEAM * TB * HID); // 512 B

    float* out = (float*)d_out;

    prep_mats<<<(HID * HID + 255) / 256, 256, 0, stream>>>(MW, MA, Wu, At, Wt, Wut,
                                                           hbuf, barc);
    lipschitz_rnn<<<NWG, NTHR, 0, stream>>>(x, bu, Wd, At, Wt, Wut, hbuf, barc, out);
}
